// MultiHeadAttention_23897198035312
// MI455X (gfx1250) — matrix operand order repaired and checked by reading, not compiled, not run on hardware
//
#include <hip/hip_runtime.h>
#include <hip/hip_bf16.h>

typedef __attribute__((ext_vector_type(16))) __bf16 v16bf;
typedef __attribute__((ext_vector_type(8)))  float  v8f;

union Frag {
  v16bf v;
  uint4 q[2];
};

__device__ __forceinline__ unsigned short f2bf(float f) {
  union { float f; unsigned u; } cv; cv.f = f;
  unsigned u = cv.u + 0x7FFFu + ((cv.u >> 16) & 1u);  // round-to-nearest-even
  return (unsigned short)(u >> 16);
}

#define HWSZ 4096
#define CCH  256
#define NH   8
#define DH   32

__device__ __forceinline__ v8f vzero8() {
  v8f z = {0.f, 0.f, 0.f, 0.f, 0.f, 0.f, 0.f, 0.f};
  return z;
}

// ---------------------------------------------------------------------------
// Kernel 1: QKV projections.  y = W @ x + b  (M=256, N=4096, K=256 per batch)
// Outputs: Qt,Kt as [b*8+h][HW][32] bf16 ; V as [b*8+h][32][HW] bf16
// Block: 128 threads (4 waves), 64x64 tile, K step 32.
// ---------------------------------------------------------------------------
__global__ __launch_bounds__(128)
void qkv_proj_kernel(const float* __restrict__ x,
                     const float* __restrict__ Wq, const float* __restrict__ bq,
                     const float* __restrict__ Wk, const float* __restrict__ bk,
                     const float* __restrict__ Wv, const float* __restrict__ bv,
                     unsigned short* __restrict__ Qt,
                     unsigned short* __restrict__ Kt,
                     unsigned short* __restrict__ Vm)
{
  const int z = blockIdx.z;           // 0..5 : proj*2 + batch
  const int p = z >> 1;
  const int b = z & 1;
  const float* W    = (p == 0) ? Wq : (p == 1) ? Wk : Wv;
  const float* bias = (p == 0) ? bq : (p == 1) ? bk : bv;
  unsigned short* dst = (p == 0) ? Qt : (p == 1) ? Kt : Vm;

  const int tileN = blockIdx.x * 64;
  const int tileM = blockIdx.y * 64;
  const int tid  = threadIdx.x;
  const int lane = tid & 31;
  const int w    = tid >> 5;
  const int half = lane >> 4;
  const int l15  = lane & 15;

  __shared__ alignas(16) unsigned short As[64 * 32];   // [m][k]
  __shared__ alignas(16) unsigned short Bs[64 * 32];   // [n][k] (transposed)

  v8f acc[4];
  for (int t = 0; t < 4; ++t) acc[t] = vzero8();

  const float* xb = x + (size_t)b * CCH * HWSZ;

  for (int k0 = 0; k0 < CCH; k0 += 32) {
    __syncthreads();
#pragma unroll
    for (int r = 0; r < 16; ++r) {                       // stage A = W tile
      int e = tid + r * 128;
      int i = e >> 5, kk = e & 31;
      As[e] = f2bf(W[(size_t)(tileM + i) * CCH + k0 + kk]);
    }
#pragma unroll
    for (int r = 0; r < 16; ++r) {                       // stage B^T = x tile
      int e = tid + r * 128;
      int kk = e >> 6, j = e & 63;
      Bs[j * 32 + kk] = f2bf(xb[(size_t)(k0 + kk) * HWSZ + tileN + j]);
    }
    __syncthreads();

    Frag a;
    {
      int row = w * 16 + l15;
      int kb = half * 8;
      a.q[0] = *reinterpret_cast<const uint4*>(&As[row * 32 + kb]);
      a.q[1] = *reinterpret_cast<const uint4*>(&As[row * 32 + kb + 16]);
    }
#pragma unroll
    for (int t = 0; t < 4; ++t) {
      Frag bf;
      int n = t * 16 + l15;
      int kb2 = half * 8;
      bf.q[0] = *reinterpret_cast<const uint4*>(&Bs[n * 32 + kb2]);
      bf.q[1] = *reinterpret_cast<const uint4*>(&Bs[n * 32 + kb2 + 16]);
      acc[t] = __builtin_amdgcn_wmma_f32_16x16x32_bf16(
          false, a.v, false, bf.v, (short)0, acc[t], false, false);
    }
  }

  const int oBase = tileM + w * 16 + 8 * half;   // 8 consecutive output channels
#pragma unroll
  for (int t = 0; t < 4; ++t) {
    int n = tileN + t * 16 + l15;
    if (p == 2) {
      // V: [b][c][n], c varies with j -> scattered b16 stores
#pragma unroll
      for (int j = 0; j < 8; ++j) {
        int o = oBase + j;
        dst[((size_t)b * CCH + o) * HWSZ + n] = f2bf(acc[t][j] + bias[o]);
      }
    } else {
      // Qt/Kt: [bh][n][dpos]; j -> 8 consecutive dpos -> one b128 store
      unsigned short h8[8];
#pragma unroll
      for (int j = 0; j < 8; ++j) h8[j] = f2bf(acc[t][j] + bias[oBase + j]);
      uint4 pk;
      pk.x = (unsigned)h8[0] | ((unsigned)h8[1] << 16);
      pk.y = (unsigned)h8[2] | ((unsigned)h8[3] << 16);
      pk.z = (unsigned)h8[4] | ((unsigned)h8[5] << 16);
      pk.w = (unsigned)h8[6] | ((unsigned)h8[7] << 16);
      size_t idx = (((size_t)(b * NH + (oBase >> 5)) * HWSZ) + n) * DH + (oBase & 31);
      *reinterpret_cast<uint4*>(&dst[idx]) = pk;
    }
  }
}

// ---------------------------------------------------------------------------
// Kernel 2: fused flash attention per (b,h).  64 q-rows per block (4 waves x 16),
// m-step of 64.  Row sums computed by a WMMA against an all-ones B matrix, so
// only the row-max needs cross-lane shuffles.
// Even/odd m-interleave: S-tile pair (even,odd) -> packed b32 LDS stores while
// keeping A-frag reads and V B-frag global loads contiguous in m.
// ---------------------------------------------------------------------------
__global__ __launch_bounds__(128)
void flash_attn_kernel(const unsigned short* __restrict__ Qt,
                       const unsigned short* __restrict__ Kt,
                       const unsigned short* __restrict__ Vm,
                       unsigned short* __restrict__ OW)
{
  const int bh   = blockIdx.y;        // 0..15
  const int b    = bh >> 3;
  const int qBlk = blockIdx.x;        // 0..63
  const int tid  = threadIdx.x;
  const int lane = tid & 31;
  const int w    = tid >> 5;
  const int half = lane >> 4;
  const int l15  = lane & 15;

  __shared__ alignas(16) unsigned short Pld[4][16 * 64];  // per-wave P tile (bf16)

  Frag qf;
  const int qRow = qBlk * 64 + w * 16 + l15;
  {
    const unsigned short* qp = Qt + ((size_t)bh * HWSZ + qRow) * DH;
    int kb = half * 16;
    qf.q[0] = *reinterpret_cast<const uint4*>(qp + kb);
    qf.q[1] = *reinterpret_cast<const uint4*>(qp + kb + 8);
  }

  Frag ones;   // bf16 1.0 broadcast: rowsum(P) = P x ones
  {
    uint4 o1; o1.x = o1.y = o1.z = o1.w = 0x3F803F80u;
    ones.q[0] = o1; ones.q[1] = o1;
  }

  v8f o0 = vzero8(), o1 = vzero8(), osum = vzero8();
  float mrow[8];
#pragma unroll
  for (int j = 0; j < 8; ++j) mrow[j] = -3.0e38f;

  const unsigned short* Kb = Kt + (size_t)bh * HWSZ * DH;   // [m][d]
  const unsigned short* Vb = Vm + (size_t)bh * DH * HWSZ;   // [d][m]
  unsigned short* pw = &Pld[w][0];
  unsigned int* pw32 = reinterpret_cast<unsigned int*>(pw);

  for (int m0 = 0; m0 < HWSZ; m0 += 64) {
    // --- S = Q K^T : 4 tiles over 64 m-columns, even/odd interleaved ------
    v8f s[4];
#pragma unroll
    for (int c = 0; c < 2; ++c) {
      Frag ke, ko;
      const unsigned short* kpe =
          Kb + (size_t)(m0 + 32 * c + 2 * l15) * DH + half * 16;
      const unsigned short* kpo = kpe + DH;              // odd row
      ke.q[0] = *reinterpret_cast<const uint4*>(kpe);
      ke.q[1] = *reinterpret_cast<const uint4*>(kpe + 8);
      ko.q[0] = *reinterpret_cast<const uint4*>(kpo);
      ko.q[1] = *reinterpret_cast<const uint4*>(kpo + 8);
      __builtin_prefetch(kpe + 64 * DH, 0, 1);           // next m-step K line
      s[2 * c] = __builtin_amdgcn_wmma_f32_16x16x32_bf16(
          false, qf.v, false, ke.v, (short)0, vzero8(), false, false);
      s[2 * c + 1] = __builtin_amdgcn_wmma_f32_16x16x32_bf16(
          false, qf.v, false, ko.v, (short)0, vzero8(), false, false);
    }

    // --- online softmax: only the MAX needs shuffles ----------------------
#pragma unroll
    for (int j = 0; j < 8; ++j) {
      float pm = fmaxf(fmaxf(s[0][j], s[1][j]), fmaxf(s[2][j], s[3][j]));
      pm = fmaxf(pm, __shfl_xor(pm, 1, 32));
      pm = fmaxf(pm, __shfl_xor(pm, 2, 32));
      pm = fmaxf(pm, __shfl_xor(pm, 4, 32));
      pm = fmaxf(pm, __shfl_xor(pm, 8, 32));
      float mn = fmaxf(mrow[j], pm);
      float sc = __expf(mrow[j] - mn);
      mrow[j] = mn;
      s[0][j] = __expf(s[0][j] - mn);
      s[1][j] = __expf(s[1][j] - mn);
      s[2][j] = __expf(s[2][j] - mn);
      s[3][j] = __expf(s[3][j] - mn);
      o0[j] *= sc;
      o1[j] *= sc;
      osum[j] *= sc;
    }

    // --- D-layout -> A-layout: packed pair stores (even col, odd col) -----
#pragma unroll
    for (int j = 0; j < 8; ++j) {
      int row = j + 8 * half;
      unsigned p0 = (unsigned)f2bf(s[0][j]) | ((unsigned)f2bf(s[1][j]) << 16);
      unsigned p1 = (unsigned)f2bf(s[2][j]) | ((unsigned)f2bf(s[3][j]) << 16);
      pw32[row * 32 + l15]      = p0;   // chunk0: ushort cols 2*l15, 2*l15+1
      pw32[row * 32 + 16 + l15] = p1;   // chunk1: cols 32+2*l15, 32+2*l15+1
    }
    asm volatile("s_wait_dscnt 0" ::: "memory");   // same-wave LDS RAW

    Frag pa[2];
    {
      int kb = half * 8;
      pa[0].q[0] = *reinterpret_cast<const uint4*>(&pw[l15 * 64 + kb]);
      pa[0].q[1] = *reinterpret_cast<const uint4*>(&pw[l15 * 64 + kb + 16]);
      pa[1].q[0] = *reinterpret_cast<const uint4*>(&pw[l15 * 64 + 32 + kb]);
      pa[1].q[1] = *reinterpret_cast<const uint4*>(&pw[l15 * 64 + 32 + kb + 16]);
    }

    // --- O += P V^T ; rowsum accumulated with ones-WMMA -------------------
#pragma unroll
    for (int c = 0; c < 2; ++c) {
      Frag v0f, v1f;
      int mo = m0 + 32 * c + half * 8;
      const unsigned short* vp0 = Vb + (size_t)l15 * HWSZ + mo;
      const unsigned short* vp1 = Vb + (size_t)(16 + l15) * HWSZ + mo;
      v0f.q[0] = *reinterpret_cast<const uint4*>(vp0);
      v0f.q[1] = *reinterpret_cast<const uint4*>(vp0 + 16);
      v1f.q[0] = *reinterpret_cast<const uint4*>(vp1);
      v1f.q[1] = *reinterpret_cast<const uint4*>(vp1 + 16);
      __builtin_prefetch(vp0 + 64, 0, 1);
      o0 = __builtin_amdgcn_wmma_f32_16x16x32_bf16(
          false, pa[c].v, false, v0f.v, (short)0, o0, false, false);
      o1 = __builtin_amdgcn_wmma_f32_16x16x32_bf16(
          false, pa[c].v, false, v1f.v, (short)0, o1, false, false);
      osum = __builtin_amdgcn_wmma_f32_16x16x32_bf16(
          false, pa[c].v, false, ones.v, (short)0, osum, false, false);
    }
  }

  // normalize (osum already per-lane: every column of D holds the row sum)
  const int h = bh & 7;
#pragma unroll
  for (int j = 0; j < 8; ++j) {
    int n = qBlk * 64 + w * 16 + j + 8 * half;
    float inv = 1.0f / osum[j];
    int c0 = h * 32 + l15;
    OW[((size_t)b * CCH + c0) * HWSZ + n]        = f2bf(o0[j] * inv);
    OW[((size_t)b * CCH + c0 + 16) * HWSZ + n]   = f2bf(o1[j] * inv);
  }
}

// ---------------------------------------------------------------------------
// Kernel 3: out = gamma * (Wo @ O + bo) + x  (f32 output)
// ---------------------------------------------------------------------------
__global__ __launch_bounds__(128)
void out_proj_kernel(const unsigned short* __restrict__ OW,
                     const float* __restrict__ Wo, const float* __restrict__ bo,
                     const float* __restrict__ x,  const float* __restrict__ gamma,
                     float* __restrict__ out)
{
  const int b = blockIdx.z;
  const int tileN = blockIdx.x * 64;
  const int tileM = blockIdx.y * 64;
  const int tid  = threadIdx.x;
  const int lane = tid & 31;
  const int w    = tid >> 5;
  const int half = lane >> 4;
  const int l15  = lane & 15;

  __shared__ alignas(16) unsigned short As[64 * 32];
  __shared__ alignas(16) unsigned short Bs[64 * 32];

  v8f acc[4];
  for (int t = 0; t < 4; ++t) acc[t] = vzero8();

  const unsigned short* ob = OW + (size_t)b * CCH * HWSZ;

  for (int k0 = 0; k0 < CCH; k0 += 32) {
    __syncthreads();
#pragma unroll
    for (int r = 0; r < 16; ++r) {
      int e = tid + r * 128;
      int i = e >> 5, kk = e & 31;
      As[e] = f2bf(Wo[(size_t)(tileM + i) * CCH + k0 + kk]);
    }
#pragma unroll
    for (int r = 0; r < 16; ++r) {
      int e = tid + r * 128;
      int kk = e >> 6, j = e & 63;
      Bs[j * 32 + kk] = ob[(size_t)(k0 + kk) * HWSZ + tileN + j];   // already bf16
    }
    __syncthreads();

    Frag a;
    {
      int row = w * 16 + l15;
      int kb = half * 8;
      a.q[0] = *reinterpret_cast<const uint4*>(&As[row * 32 + kb]);
      a.q[1] = *reinterpret_cast<const uint4*>(&As[row * 32 + kb + 16]);
    }
#pragma unroll
    for (int t = 0; t < 4; ++t) {
      Frag bf;
      int n = t * 16 + l15;
      int kb2 = half * 8;
      bf.q[0] = *reinterpret_cast<const uint4*>(&Bs[n * 32 + kb2]);
      bf.q[1] = *reinterpret_cast<const uint4*>(&Bs[n * 32 + kb2 + 16]);
      acc[t] = __builtin_amdgcn_wmma_f32_16x16x32_bf16(
          false, a.v, false, bf.v, (short)0, acc[t], false, false);
    }
  }

  const float g = gamma[0];
#pragma unroll
  for (int t = 0; t < 4; ++t) {
    int n = tileN + t * 16 + l15;
#pragma unroll
    for (int j = 0; j < 8; ++j) {
      int o = tileM + w * 16 + j + 8 * half;
      size_t idx = ((size_t)b * CCH + o) * HWSZ + n;
      out[idx] = g * (acc[t][j] + bo[o]) + x[idx];
    }
  }
}

// ---------------------------------------------------------------------------
extern "C" void kernel_launch(void* const* d_in, const int* in_sizes, int n_in,
                              void* d_out, int out_size, void* d_ws, size_t ws_size,
                              hipStream_t stream) {
  (void)in_sizes; (void)n_in; (void)out_size; (void)ws_size;
  const float* x     = (const float*)d_in[0];
  const float* Wq    = (const float*)d_in[1];
  const float* bq    = (const float*)d_in[2];
  const float* Wk    = (const float*)d_in[3];
  const float* bk    = (const float*)d_in[4];
  const float* Wv    = (const float*)d_in[5];
  const float* bv    = (const float*)d_in[6];
  const float* Wo    = (const float*)d_in[7];
  const float* bo    = (const float*)d_in[8];
  const float* gamma = (const float*)d_in[9];
  float* out = (float*)d_out;

  // Workspace layout (bf16 as ushort): Qt | Kt | V | O, each 2*8*4096*32 elems (4 MB)
  const size_t SEG = (size_t)2 * NH * HWSZ * DH;   // 2,097,152 elements
  unsigned short* Qt = (unsigned short*)d_ws;
  unsigned short* Kt = Qt + SEG;
  unsigned short* Vm = Kt + SEG;
  unsigned short* OW = Vm + SEG;

  qkv_proj_kernel<<<dim3(HWSZ / 64, CCH / 64, 6), dim3(128), 0, stream>>>(
      x, Wq, bq, Wk, bk, Wv, bv, Qt, Kt, Vm);

  flash_attn_kernel<<<dim3(HWSZ / 64, 16), dim3(128), 0, stream>>>(
      Qt, Kt, Vm, OW);

  out_proj_kernel<<<dim3(HWSZ / 64, CCH / 64, 2), dim3(128), 0, stream>>>(
      OW, Wo, bo, x, gamma, out);
}
